// VIN_21251498180907
// MI455X (gfx1250) — compile-verified, matrix operand order checked by reading, not run on hardware
//
#include <hip/hip_runtime.h>
#include <hip/hip_bf16.h>
#include <math.h>

typedef __attribute__((ext_vector_type(16))) _Float16 v16h;
typedef __attribute__((ext_vector_type(8)))  float    v8f;

#define HCH 150
#define NPAD 160   // 10 N-tiles of 16
#define QCH 10

// ---------------------------------------------------------------------------
// Kernel 1: fused conv1 (2->150ch 3x3 + bias) and conv2 (150->1ch 3x3).
// One block per (16-row strip, image). h is computed one row at a time via
// WMMA implicit GEMM (M=64 px, N=160 ch, K=19) and contracted with Wr in LDS.
// ---------------------------------------------------------------------------
__global__ void __launch_bounds__(256)
vin_r_fused_kernel(const float* __restrict__ obs, const float* __restrict__ Wh,
                   const float* __restrict__ bh,  const float* __restrict__ Wr,
                   float* __restrict__ r_out) {
    __shared__ _Float16 obs_s[2][20][66];   // rows r0-2 .. r0+17, cols zero-padded
    __shared__ float    h_s[NPAD][68];      // one h row; x stored at col x+1, cols 0/65 zero
    __shared__ float    wr_s[HCH][9];
    __shared__ float    racc[16][64];

    const int b    = blockIdx.y;
    const int r0   = blockIdx.x * 16;
    const int t    = threadIdx.x;
    const int lane = t & 31;               // wave32
    const int wv   = t >> 5;               // 8 waves
    const int mtile = wv >> 1;             // 4 pixel tiles of 16
    const int nh    = wv & 1;              // 2 halves of 10 channel tiles

    // Wr -> LDS
    for (int i = t; i < HCH * 9; i += 256) wr_s[i / 9][i % 9] = Wr[i];
    // zero r accumulator and the padded h columns (written once, never touched again)
    for (int i = t; i < 16 * 64; i += 256) racc[i >> 6][i & 63] = 0.0f;
    for (int i = t; i < NPAD; i += 256) { h_s[i][0] = 0.0f; h_s[i][65] = 0.0f; }
    // obs strip -> LDS (f32 -> f16), zero padding outside the image
    for (int i = t; i < 2 * 20 * 66; i += 256) {
        int ci  = i / (20 * 66);
        int rem = i % (20 * 66);
        int rl  = rem / 66;
        int cl  = rem % 66;
        int gy  = r0 - 2 + rl;
        int gx  = cl - 1;
        float v = 0.0f;
        if (gy >= 0 && gy < 64 && gx >= 0 && gx < 64)
            v = obs[(((size_t)b * 2 + ci) * 64 + gy) * 64 + gx];
        obs_s[ci][rl][cl] = (_Float16)v;
    }

    // Persistent B fragments: Wh columns for this wave's 5 N-tiles.
    // K = ci*9 + tap for K<18 ; K==18 carries the bias row (A supplies 1.0).
    const int nlocal = lane & 15;
    const int khalf  = (lane < 16) ? 0 : 8;
    v16h bfrag[5];
    #pragma unroll
    for (int f = 0; f < 5; ++f) {
        int ch = (nh * 5 + f) * 16 + nlocal;
        #pragma unroll
        for (int e = 0; e < 16; ++e) {
            int K = e + ((e < 8) ? 0 : 8) + khalf;
            float v = 0.0f;
            if (ch < HCH) {
                if (K < 18)       v = Wh[(ch * 2 + (K / 9)) * 9 + (K % 9)];
                else if (K == 18) v = bh[ch];
            }
            bfrag[f][e] = (_Float16)v;
        }
    }
    __syncthreads();

    const _Float16* zslot = &obs_s[0][0][0];   // guaranteed zero (padded corner)

    // Sweep the 18 h rows feeding this 16-row r strip.
    for (int hi = 0; hi < 18; ++hi) {
        int hr = r0 - 1 + hi;
        if (hr >= 0 && hr < 64) {           // uniform branch; EXEC stays full
            // A fragment: 16 pixels x K=32 im2col. All loads unconditional:
            // invalid K slots read the zero corner via a branchless address select.
            const int m  = lane & 15;
            const int x  = mtile * 16 + m;
            const int rb = hr - r0 + 1;
            v16h afrag;
            #pragma unroll
            for (int e = 0; e < 16; ++e) {
                const int Klo = e + ((e < 8) ? 0 : 8);   // lanes 0-15
                const int Khi = Klo + 8;                 // lanes 16-31
                const _Float16* plo = (Klo < 18)
                    ? &obs_s[Klo / 9][rb + ((Klo % 9) / 3)][x + ((Klo % 9) % 3)]
                    : zslot;
                const _Float16* phi = (Khi < 18)
                    ? &obs_s[Khi / 9][rb + ((Khi % 9) / 3)][x + ((Khi % 9) % 3)]
                    : zslot;
                _Float16 v = *((lane < 16) ? plo : phi);
                if (Klo == 18) v = (lane < 16) ? (_Float16)1.0f : v;  // bias row
                afrag[e] = v;
            }
            // 5 WMMAs: h tile (16 px x 16 ch), K=19 fits in a single x32 op
            v8f cf[5];
            #pragma unroll
            for (int f = 0; f < 5; ++f) {
                v8f cz = {};
                cf[f] = __builtin_amdgcn_wmma_f32_16x16x32_f16(
                    false, afrag, false, bfrag[f], (short)0, cz, false, false);
            }
            // Scatter C (M = v + lanehalf*8, N = lane&15) into padded h row
            #pragma unroll
            for (int f = 0; f < 5; ++f) {
                int ch = (nh * 5 + f) * 16 + nlocal;
                #pragma unroll
                for (int vv = 0; vv < 8; ++vv) {
                    int xx = mtile * 16 + vv + ((lane < 16) ? 0 : 8);
                    h_s[ch][xx + 1] = cf[f][vv];
                }
            }
            __syncthreads();

            // Contract this h row with Wr into the r accumulator.
            // h row hr feeds output rows hr-1, hr, hr+1 (ky = hr-y+1).
            // Fully branchless: h_s is zero-padded in x.
            if (t < 192) {
                int y  = hr - 1 + (t >> 6);
                int x2 = t & 63;
                if (y >= r0 && y < r0 + 16) {
                    int ky3 = (hr - y + 1) * 3;
                    float acc = 0.0f;
                    #pragma unroll 2
                    for (int c = 0; c < HCH; ++c) {
                        acc = fmaf(wr_s[c][ky3 + 0], h_s[c][x2 + 0],
                              fmaf(wr_s[c][ky3 + 1], h_s[c][x2 + 1],
                              fmaf(wr_s[c][ky3 + 2], h_s[c][x2 + 2], acc)));
                    }
                    racc[y - r0][x2] += acc;
                }
            }
            __syncthreads();
        }
    }

    // Flush the strip of r to the workspace.
    for (int i = t; i < 16 * 64; i += 256)
        r_out[((size_t)b * 64 + r0 + (i >> 6)) * 64 + (i & 63)] = racc[i >> 6][i & 63];
}

// ---------------------------------------------------------------------------
// Kernel 2: value-iteration loop, one block per image, everything in LDS.
// conv(r, w_i2q) is iteration-invariant: precomputed once into registers
// (qi), so each of the k steps only does conv(v, w_v2q) + max.
// 1024 threads x 4 contiguous pixels each.
// ---------------------------------------------------------------------------
__global__ void __launch_bounds__(1024)
vin_vi_kernel(const float* __restrict__ r_in, const float* __restrict__ w_i2q,
              const float* __restrict__ w_v2q, const float* __restrict__ Wfc,
              const int* __restrict__ s1, const int* __restrict__ s2,
              const int* __restrict__ kptr, float* __restrict__ out) {
    __shared__ float r_s[66][66];    // zero-padded reward map
    __shared__ float v_s[66][66];    // zero-padded value map
    __shared__ float wi_s[QCH][9];
    __shared__ float wv_s[QCH][9];
    __shared__ float qatt_s[QCH];

    const int b  = blockIdx.x;
    const int t  = threadIdx.x;
    const int K  = kptr[0];
    const int y  = t >> 4;            // 0..63
    const int x0 = (t & 15) << 2;     // 0,4,...,60

    if (t == 0) __builtin_prefetch(&r_in[(size_t)b * 4096], 0, 1);

    for (int i = t; i < QCH * 9; i += 1024) { wi_s[i / 9][i % 9] = w_i2q[i];
                                              wv_s[i / 9][i % 9] = w_v2q[i]; }
    for (int i = t; i < 66 * 66; i += 1024) { r_s[i / 66][i % 66] = 0.0f;
                                              v_s[i / 66][i % 66] = 0.0f; }
    __syncthreads();
    for (int i = t; i < 4096; i += 1024)
        r_s[(i >> 6) + 1][(i & 63) + 1] = r_in[(size_t)b * 4096 + i];
    __syncthreads();

    // qi[i][a] = conv(r, w_i2q)[a] at pixel (y, x0+i): iteration-invariant.
    float qi[4][QCH];
    {
        float rr[3][6];
        #pragma unroll
        for (int ky = 0; ky < 3; ++ky)
            #pragma unroll
            for (int j = 0; j < 6; ++j) rr[ky][j] = r_s[y + ky][x0 + j];
        #pragma unroll
        for (int i = 0; i < 4; ++i)
            #pragma unroll
            for (int a = 0; a < QCH; ++a) {
                float q = 0.0f;
                #pragma unroll
                for (int ky = 0; ky < 3; ++ky)
                    #pragma unroll
                    for (int kx = 0; kx < 3; ++kx)
                        q = fmaf(wi_s[a][ky * 3 + kx], rr[ky][i + kx], q);
                qi[i][a] = q;
            }
    }
    // v0 = max_a qi
    float nv[4];
    #pragma unroll
    for (int i = 0; i < 4; ++i) {
        float m = qi[i][0];
        #pragma unroll
        for (int a = 1; a < QCH; ++a) m = fmaxf(m, qi[i][a]);
        nv[i] = m;
    }
    #pragma unroll
    for (int i = 0; i < 4; ++i) v_s[y + 1][x0 + 1 + i] = nv[i];
    __syncthreads();

    // k value-iteration steps: only the v-half of the conv remains.
    for (int it = 0; it < K; ++it) {
        float vv[3][6];
        #pragma unroll
        for (int ky = 0; ky < 3; ++ky)
            #pragma unroll
            for (int j = 0; j < 6; ++j) vv[ky][j] = v_s[y + ky][x0 + j];
        #pragma unroll
        for (int i = 0; i < 4; ++i) {
            float m = -INFINITY;
            #pragma unroll
            for (int a = 0; a < QCH; ++a) {
                float q = qi[i][a];
                #pragma unroll
                for (int ky = 0; ky < 3; ++ky)
                    #pragma unroll
                    for (int kx = 0; kx < 3; ++kx)
                        q = fmaf(wv_s[a][ky * 3 + kx], vv[ky][i + kx], q);
                m = fmaxf(m, q);
            }
            nv[i] = m;
        }
        __syncthreads();
        #pragma unroll
        for (int i = 0; i < 4; ++i) v_s[y + 1][x0 + 1 + i] = nv[i];
        __syncthreads();
    }

    // Final q at the attended cell, then FC to 8 logits.
    if (t == 0) {
        int yy = s1[b], xx = s2[b];
        for (int a = 0; a < QCH; ++a) {
            float q = 0.0f;
            #pragma unroll
            for (int tap = 0; tap < 9; ++tap) {
                q = fmaf(wi_s[a][tap], r_s[yy + tap / 3][xx + tap % 3], q);
                q = fmaf(wv_s[a][tap], v_s[yy + tap / 3][xx + tap % 3], q);
            }
            qatt_s[a] = q;
        }
    }
    __syncthreads();
    if (t < 8) {
        float acc = 0.0f;
        for (int a = 0; a < QCH; ++a) acc = fmaf(Wfc[t * QCH + a], qatt_s[a], acc);
        out[(size_t)b * 8 + t] = acc;
    }
}

// ---------------------------------------------------------------------------
extern "C" void kernel_launch(void* const* d_in, const int* in_sizes, int n_in,
                              void* d_out, int out_size, void* d_ws, size_t ws_size,
                              hipStream_t stream) {
    const int*   s1    = (const int*)d_in[0];
    const int*   s2    = (const int*)d_in[1];
    const float* obs   = (const float*)d_in[2];
    const int*   kptr  = (const int*)d_in[3];
    const float* Wh    = (const float*)d_in[4];
    const float* bh    = (const float*)d_in[5];
    const float* Wr    = (const float*)d_in[6];
    const float* w_i2q = (const float*)d_in[7];
    const float* w_v2q = (const float*)d_in[8];
    const float* Wfc   = (const float*)d_in[9];
    float* logits = (float*)d_out;
    float* r_ws   = (float*)d_ws;   // 128*64*64 floats = 2 MB reward maps

    dim3 g1(4, 128);   // 4 row strips x 128 images
    vin_r_fused_kernel<<<g1, 256, 0, stream>>>(obs, Wh, bh, Wr, r_ws);
    vin_vi_kernel<<<128, 1024, 0, stream>>>(r_ws, w_i2q, w_v2q, Wfc, s1, s2, kptr, logits);
}
